// FFTSpatiallyVariantBlur_75316546502865
// MI455X (gfx1250) — matrix operand order repaired and checked by reading, not compiled, not run on hardware
//
#include <hip/hip_runtime.h>
#include <hip/hip_bf16.h>

// CDNA5 / gfx1250: wave32, v_wmma_f32_16x16x32_bf16, f32 accumulate.
typedef __attribute__((ext_vector_type(16))) __bf16 v16bf;
typedef __attribute__((ext_vector_type(8)))  __bf16 bf16x8;
typedef __attribute__((ext_vector_type(2)))  __bf16 bf16x2;
typedef __attribute__((ext_vector_type(8)))  float  v8f;

#define TILE 32
#define RAD  15
#define KS   31
#define IN   62            // TILE + 2*RAD valid extent
#define XROW 88            // x_tile row stride (bf16): 176B = 11*16, 44m%64 distinct
#define HROW 72            // Ht row stride (bf16): 144B = 9*16, 36m%64 distinct
#define NB   8
#define GBO  64            // zero-pad offset into gb

__device__ __forceinline__ v16bf cat8(bf16x8 lo, bf16x8 hi) {
  return __builtin_shufflevector(lo, hi, 0,1,2,3,4,5,6,7,8,9,10,11,12,13,14,15);
}

__global__ __launch_bounds__(256)
void fft_svb_kernel(const float* __restrict__ image,
                    const float* __restrict__ coc,
                    const float* __restrict__ w_sigma,
                    const float* __restrict__ b_sigma,
                    float* __restrict__ out)
{
  __shared__ __bf16 x_tile[64 * XROW];        // 11.0 KB zero-padded input tile
  __shared__ __bf16 Ht[NB * 32 * HROW];       // 36.0 KB per-base H panel, transposed
  __shared__ float  g[NB][32];                // raw taps
  __shared__ __bf16 gb[NB * 128];             // zero-padded normalized taps (bf16)
  __shared__ float  ginv[NB];
  __shared__ float  sig[TILE * TILE];
  __shared__ float  invden[TILE * TILE];
  __shared__ float  acc_tile[TILE * TILE];

  const int tid  = threadIdx.x;
  const int lane = tid & 31;
  const int wave = tid >> 5;                  // one wave per PSF base
  const int h    = lane >> 4;
  const int m    = lane & 15;

  const int x0 = blockIdx.x * TILE;
  const int y0 = blockIdx.y * TILE;
  const int bc = blockIdx.z;                  // b*3 + c
  const int b  = bc / 3;

  const float ws = w_sigma[0];
  const float bs = b_sigma[0];
  const float sigma_w = 0.2f + (float)wave * (11.8f / 7.0f);

  // ---- stage 0: fills (no barriers needed between independent arrays) ------
  // input tile as bf16, zero-padded to 64 x 88 (pairs -> 4B stores)
  for (int i = tid; i < 64 * (XROW / 2); i += 256) {
    int r  = i / (XROW / 2);
    int c  = (i - r * (XROW / 2)) * 2;
    float v0 = 0.f, v1 = 0.f;
    int gy = y0 + r - RAD;
    if (r < IN && gy >= 0 && gy < 512) {
      int gx = x0 + c - RAD;
      const float* src = &image[((size_t)bc * 512 + gy) * 512];
      if (c     < IN && gx     >= 0 && gx     < 512) v0 = src[gx];
      if (c + 1 < IN && gx + 1 >= 0 && gx + 1 < 512) v1 = src[gx + 1];
    }
    bf16x2 pv; pv.x = (__bf16)v0; pv.y = (__bf16)v1;
    *(bf16x2*)&x_tile[r * XROW + c] = pv;
  }
  // zero-pad taps array
  for (int i = tid; i < NB * 128; i += 256) gb[i] = (__bf16)0.f;
  // raw (unnormalized) separable taps
  {
    int k = tid >> 5, t = tid & 31;
    float sk = 0.2f + (float)k * (11.8f / 7.0f);
    float v = 0.f;
    if (t < KS) {
      float dx = (float)(t - RAD);
      v = expf(-dx * dx / (2.f * sk * sk + 1e-9f));
    }
    g[k][t] = v;
  }
  // sigma map + weight normalizer + accumulator clear
  for (int p = tid; p < TILE * TILE; p += 256) {
    int y = p >> 5, x = p & 31;
    acc_tile[p] = 0.f;
    float cv = coc[((size_t)b * 512 + (y0 + y)) * 512 + (x0 + x)];
    float z  = cv * ws + bs;
    float sp = (z > 20.f) ? z : log1pf(expf(z));
    sp = fminf(fmaxf(sp, 0.2f), 12.0f);
    sig[p] = sp;
    float den = 0.f;
#pragma unroll
    for (int k = 0; k < NB; ++k) {
      float sk = 0.2f + (float)k * (11.8f / 7.0f);
      float d  = sp - sk;
      den += expf(-0.5f * d * d);
    }
    invden[p] = 1.f / (den + 1e-9f);
  }
  __syncthreads();
  if (tid < NB) {                 // 1D/S so (1D ⊗ 1D)/S^2 == 2D/sum2d
    float s = 0.f;
    for (int t = 0; t < KS; ++t) s += g[tid][t];
    ginv[tid] = 1.f / s;
  }
  __syncthreads();
  {                               // normalized taps -> zero-padded bf16
    int k = tid >> 5, t = tid & 31;
    if (t < KS) gb[k * 128 + GBO + t] = (__bf16)(g[k][t] * ginv[k]);
  }
  __syncthreads();

  // ---- stage 1: horizontal Toeplitz matmul (per-wave base) -----------------
  // H[r][c0+j] = sum_kk X[r][c0+kk] * g[kk-j], K 46 -> 64 (2 x K32)
  const __bf16* gbw = &gb[wave * 128];
  __bf16* htw = &Ht[wave * 32 * HROW];

  v16bf Bh[2];
#pragma unroll
  for (int ch = 0; ch < 2; ++ch) {
    int b0 = GBO + ch * 32 + h * 8 - m;       // B[kb][n=m] = g[kb - m]
    v16bf f;
#pragma unroll
    for (int e = 0; e < 16; ++e) f[e] = gbw[b0 + (e < 8 ? e : e + 8)];
    Bh[ch] = f;
  }
#pragma unroll
  for (int rb = 0; rb < 4; ++rb) {
#pragma unroll
    for (int cb = 0; cb < 2; ++cb) {
      v8f acc = {};
#pragma unroll
      for (int ch = 0; ch < 2; ++ch) {
        // A runs: e<8 -> cols base+h*8.., e>=8 -> cols base+16+h*8..
        const __bf16* ap = &x_tile[(rb * 16 + m) * XROW + cb * 16 + ch * 32 + h * 8];
        v16bf A = cat8(*(const bf16x8*)ap, *(const bf16x8*)(ap + 16));
        acc = __builtin_amdgcn_wmma_f32_16x16x32_bf16(
                false, A, false, Bh[ch], (short)0, acc, false, false);
      }
      bf16x8 d8;
#pragma unroll
      for (int r = 0; r < 8; ++r) d8[r] = (__bf16)acc[r];
      // Ht[col][row]: col = cb*16+m (n), rows rb*16+8h .. +7
      *(bf16x8*)&htw[(cb * 16 + m) * HROW + rb * 16 + 8 * h] = d8;
    }
  }
  // same-wave LDS RAW on Ht: DS ops are in-order within a wave

  // ---- stage 2: vertical Toeplitz matmul + weighted reduction --------------
  // Out[y][x] = sum_i g[i-y] * H[i][x], K 62 -> 64
#pragma unroll
  for (int yb = 0; yb < 2; ++yb) {
    v16bf Av[2];
#pragma unroll
    for (int ch = 0; ch < 2; ++ch) {
      int base = GBO + ch * 32 + h * 8 - (yb * 16 + m);   // T[y][i] = g[i-y]
      v16bf f;
#pragma unroll
      for (int e = 0; e < 16; ++e) f[e] = gbw[base + (e < 8 ? e : e + 8)];
      Av[ch] = f;
    }
#pragma unroll
    for (int xb = 0; xb < 2; ++xb) {
      v8f acc = {};
#pragma unroll
      for (int ch = 0; ch < 2; ++ch) {
        // B: Ht column (xb*16+m), rows h*8..+7 and 16+h*8..+7 of the chunk
        const __bf16* bp = &htw[(xb * 16 + m) * HROW + ch * 32 + h * 8];
        v16bf B = cat8(*(const bf16x8*)bp, *(const bf16x8*)(bp + 16));
        acc = __builtin_amdgcn_wmma_f32_16x16x32_bf16(
                false, Av[ch], false, B, (short)0, acc, false, false);
      }
#pragma unroll
      for (int r = 0; r < 8; ++r) {
        int y = yb * 16 + r + 8 * h;
        int x = xb * 16 + m;
        int p = y * TILE + x;
        float d = sig[p] - sigma_w;
        float w = expf(-0.5f * d * d) * invden[p];
        atomicAdd(&acc_tile[p], w * acc[r]);  // ds_add_f32 across the 8 bases
      }
    }
  }
  __syncthreads();

  // ---- stage 3: write output ----------------------------------------------
  for (int p = tid; p < TILE * TILE; p += 256) {
    int y = p >> 5, x = p & 31;
    out[((size_t)bc * 512 + (y0 + y)) * 512 + (x0 + x)] = acc_tile[p];
  }
}

extern "C" void kernel_launch(void* const* d_in, const int* in_sizes, int n_in,
                              void* d_out, int out_size, void* d_ws, size_t ws_size,
                              hipStream_t stream) {
  const float* image = (const float*)d_in[0];   // (8,3,512,512) f32
  const float* coc   = (const float*)d_in[1];   // (8,1,512,512) f32
  // d_in[2] = psf_params: unused by the reference computation
  const float* wsg   = (const float*)d_in[3];   // scalar
  const float* bsg   = (const float*)d_in[4];   // scalar
  float* out = (float*)d_out;                   // (8,3,512,512) f32

  dim3 grid(512 / TILE, 512 / TILE, 8 * 3);
  fft_svb_kernel<<<grid, 256, 0, stream>>>(image, coc, wsg, bsg, out);

  (void)in_sizes; (void)n_in; (void)out_size; (void)d_ws; (void)ws_size;
}
